// Physics_Cross_Attention_plus_16836271800867
// MI455X (gfx1250) — matrix operand order repaired and checked by reading, not compiled, not run on hardware
//
#include <hip/hip_runtime.h>
#include <hip/hip_bf16.h>
#include <cstdint>

// ---------------------------------------------------------------------------
// Physics Cross Attention for MI455X (gfx1250), wave32 + WMMA f16 + TDM.
// Memory-bound (~300MB @ 23.3TB/s => ~13us floor); all GEMMs via
// v_wmma_f32_16x16x32_f16, fp16 intermediates, fused gumbel-softmax routing.
// B-operand tiles staged by the Tensor Data Mover (tensor_load_to_lds with
// LDS padding producing the 40-half row stride); A tiles staged with 128-bit
// global loads + b64/b128 LDS stores.
// ---------------------------------------------------------------------------

typedef __attribute__((ext_vector_type(16))) _Float16 v16h;
typedef __attribute__((ext_vector_type(8)))  float    v8f;
typedef unsigned int u32x4 __attribute__((ext_vector_type(4)));
typedef int          i32x8 __attribute__((ext_vector_type(8)));
typedef int          i32x4 __attribute__((ext_vector_type(4)));

#define H_ 8
#define D_ 64
#define G_ 64
#define DIM_ 512
#define N_TOK 8192
#define M_TOK 1024
#define B_ 4

#define LDSL 40   // LDS row stride in halves (80B: b128-aligned, pair-read ok)

struct alignas(8)  H4 { _Float16 h[4]; };
struct alignas(16) H8 { _Float16 h[8]; };

__device__ __forceinline__ float gelu_f(float x) {
  return 0.5f * x * (1.0f + erff(x * 0.70710678118654752f));
}

// LDS byte offset of a generic pointer to __shared__ (flat->LDS: addr[31:0])
__device__ __forceinline__ unsigned lds_off(const void* p) {
  return (unsigned)(unsigned long long)p;
}

// Issue one TDM 2D tile load: tile_dim0 halves x tile_dim1 lines from
// src (line stride = ld halves) into LDS at ldsByteOff, padding 16B after
// every 64B so LDS line stride is 40 halves.
__device__ __forceinline__ void tdm_load_tile_b(const _Float16* src,
                                                unsigned ldsByteOff,
                                                unsigned ld) {
  unsigned long long ga = (unsigned long long)(const void*)src;
  u32x4 g0;
  g0[0] = 1u;                                   // count=1, user descriptor
  g0[1] = ldsByteOff;                           // lds_addr (bytes)
  g0[2] = (unsigned)ga;                         // global_addr[31:0]
  g0[3] = (unsigned)((ga >> 32) & 0x01FFFFFFu) | (2u << 30);  // [56:32]|type=2
  const unsigned td0 = ld;                      // tensor_dim0 (data_size units)
  const unsigned td1 = 1u << 20;                // tensor_dim1 (rows, generous)
  i32x8 g1;
  // data_size=2B (code 1), pad_enable, pad_interval=16 DWORDs (code 3),
  // pad_amount=4 DWORDs (code 3)
  g1[0] = (int)((1u << 16) | (1u << 20) | (3u << 22) | (3u << 25));
  g1[1] = (int)((td0 & 0xFFFFu) << 16);         // tensor_dim0[15:0] @bit48
  g1[2] = (int)((td0 >> 16) | ((td1 & 0xFFFFu) << 16));  // dim0 hi | dim1 lo
  g1[3] = (int)((td1 >> 16) | (32u << 16));     // dim1 hi | tile_dim0=32
  g1[4] = 64;                                   // tile_dim1=64, tile_dim2=0
  g1[5] = (int)td0;                             // tensor_dim0_stride lo32
  g1[6] = 0;                                    // stride hi16 | dim1_stride lo
  g1[7] = 0;
  i32x4 g2 = {0, 0, 0, 0};
  i32x4 g3 = {0, 0, 0, 0};
  i32x8 g4 = {0, 0, 0, 0, 0, 0, 0, 0};
  __builtin_amdgcn_tensor_load_to_lds(g0, g1, g2, g3, g4, 0);
}

// ---------------------------------------------------------------------------
// Generic WMMA GEMM: C[M,N] = A[M,K] * B[K,N] (+bias), batched, K-splittable.
// Block tile 64x64, 256 threads = 8 waves, each wave: 16 rows x 32 cols
// (two v_wmma_f32_16x16x32_f16). LDS staging: lds_a[m][k], lds_b[n][k] so
// both fragments read contiguous K-pairs per ISA 7.12.2 layouts.
//
// AMODE: 0=A f32   1=A f16   2=A f32 * 1/(rowScale+1e-5)   3=A f16 transposed
// BMODE: 0=B f16 row-major [K][N] (manual transpose staging)
//        1=B f16 col-major [N][K] (TDM tile DMA)
// EPI:   0=f16 [M,N]           1=f16 transposed C[n*ldc+m]
//        2=f16 scatter (b,n)x(h,d) -> [b,h,n,d], +bias
//        3=f32 [M,N] + bias    4=f32 atomicAdd (K-split)
//        5=f16 combine scatter bat=(b,h), (n)x(d) -> [b,n,h*64+d]
// ---------------------------------------------------------------------------
struct GP {
  const void* A; const void* B; const float* bias; const float* rowScale;
  void* C;
  int M, N, K;
  int lda, ldb, ldc;
  long long sA, sB, sC;     // batch strides in elements
  int ksplit, Np, Hdim;
};

template <int AMODE, int BMODE, int EPI>
__global__ __launch_bounds__(256) void gemm_wmma(GP p) {
  __shared__ _Float16 lds_a[64 * LDSL];
  __shared__ _Float16 lds_b[64 * LDSL];

  const int tid  = threadIdx.x;
  const int wid  = tid >> 5;
  const int lane = tid & 31;

  const int bat  = (int)(blockIdx.z) / p.ksplit;
  const int kch  = (int)(blockIdx.z) % p.ksplit;
  const int Kper = p.K / p.ksplit;
  const int k_begin = kch * Kper, k_end = k_begin + Kper;

  const int rowBase = blockIdx.y * 64;
  const int colBase = blockIdx.x * 64;
  const int rowTile  = (wid & 3) * 16;
  const int colTile0 = (wid >> 2) * 32;

  const _Float16* Bh = (const _Float16*)p.B + (long long)bat * p.sB;
  const unsigned ldsbOff = lds_off(&lds_b[0]);

  v8f acc0 = {};
  v8f acc1 = {};

  for (int k0 = k_begin; k0 < k_end; k0 += 32) {
    // ---------------- stage A tile 64x32 into lds_a[m][k] ----------------
    if constexpr (AMODE == 0 || AMODE == 2) {
      // fp32 row-major: 2 x float4 per thread (loads issued together)
      const float* Af = (const float*)p.A + (long long)bat * p.sA;
      const int r0 = tid >> 3, kk = (tid & 7) * 4;
      const int r1 = r0 + 32;
      float4 v0 = *(const float4*)(Af + (long long)(rowBase + r0) * p.lda + k0 + kk);
      float4 v1 = *(const float4*)(Af + (long long)(rowBase + r1) * p.lda + k0 + kk);
      float s0 = 1.f, s1 = 1.f;
      if constexpr (AMODE == 2) {
        s0 = 1.0f / (p.rowScale[bat * p.M + rowBase + r0] + 1e-5f);
        s1 = 1.0f / (p.rowScale[bat * p.M + rowBase + r1] + 1e-5f);
      }
      H4 a0, a1;
      a0.h[0] = (_Float16)(v0.x * s0); a0.h[1] = (_Float16)(v0.y * s0);
      a0.h[2] = (_Float16)(v0.z * s0); a0.h[3] = (_Float16)(v0.w * s0);
      a1.h[0] = (_Float16)(v1.x * s1); a1.h[1] = (_Float16)(v1.y * s1);
      a1.h[2] = (_Float16)(v1.z * s1); a1.h[3] = (_Float16)(v1.w * s1);
      *(H4*)&lds_a[r0 * LDSL + kk] = a0;
      *(H4*)&lds_a[r1 * LDSL + kk] = a1;
    } else if constexpr (AMODE == 1) {
      // f16 row-major: one 16B load + one b128 LDS store per thread
      const _Float16* Ah = (const _Float16*)p.A + (long long)bat * p.sA;
      const int r = tid >> 2, kk8 = (tid & 3) * 8;
      uint4 v = *(const uint4*)(Ah + (long long)(rowBase + r) * p.lda + k0 + kk8);
      *(uint4*)&lds_a[r * LDSL + kk8] = v;
    } else { // AMODE 3: A^T, element [g][n] = Asrc[n*lda + g]; g contiguous
      const _Float16* Ah = (const _Float16*)p.A + (long long)bat * p.sA;
      const int nn = tid >> 3, g0 = (tid & 7) * 8;
      H8 v = *(const H8*)(Ah + (long long)(k0 + nn) * p.lda + rowBase + g0);
#pragma unroll
      for (int e = 0; e < 8; e++) lds_a[(g0 + e) * LDSL + nn] = v.h[e];
    }

    // ---------------- stage B tile 32x64 into lds_b[n][k] ----------------
    if constexpr (BMODE == 0) {
      // row-major [K][N]: 16B load along n, transpose on LDS write
      const int kk = tid >> 3, n0 = (tid & 7) * 8;
      H8 v = *(const H8*)(Bh + (long long)(k0 + kk) * p.ldb + colBase + n0);
#pragma unroll
      for (int e = 0; e < 8; e++) lds_b[(n0 + e) * LDSL + kk] = v.h[e];
    } else {
      // col-major [N][K]: Tensor Data Mover DMA, wave 0 issues one tile op
      if (wid == 0) {
        tdm_load_tile_b(Bh + (long long)colBase * p.ldb + k0, ldsbOff,
                        (unsigned)p.ldb);
        __builtin_amdgcn_s_wait_tensorcnt(0);
      }
    }

    // prefetch next A tile (global_prefetch_b8 path)
    if (k0 + 32 < k_end) {
      if constexpr (AMODE == 0 || AMODE == 2) {
        const float* Af = (const float*)p.A + (long long)bat * p.sA;
        __builtin_prefetch(Af + (long long)(rowBase + (tid >> 2)) * p.lda + k0 + 32, 0, 1);
      } else {
        const _Float16* Ah = (const _Float16*)p.A + (long long)bat * p.sA;
        __builtin_prefetch(Ah + (long long)(rowBase + (tid >> 2)) * p.lda + k0 + 32, 0, 1);
      }
    }
    __syncthreads();

    // ---- build fragments from LDS per WMMA register layouts ----
    union Frag { v16h v; unsigned u[8]; } fa, fb0, fb1;
    {
      const int r   = rowTile + (lane & 15);
      const int kba = (lane >> 4) * 16;           // A: lane-hi group of 16 K
      const unsigned* pa = (const unsigned*)&lds_a[r * LDSL];
#pragma unroll
      for (int e = 0; e < 8; e++) {
        int kk = kba + 2 * e;
        fa.u[e] = pa[kk >> 1];
      }
      const int kbb = (lane >> 4) * 16;           // B: lane-hi group of 16 K
      const unsigned* pb0 = (const unsigned*)&lds_b[(colTile0 + (lane & 15)) * LDSL];
      const unsigned* pb1 = (const unsigned*)&lds_b[(colTile0 + 16 + (lane & 15)) * LDSL];
#pragma unroll
      for (int e = 0; e < 8; e++) {
        int kk = kbb + 2 * e;
        fb0.u[e] = pb0[kk >> 1];
        fb1.u[e] = pb1[kk >> 1];
      }
    }
    acc0 = __builtin_amdgcn_wmma_f32_16x16x32_f16(false, fa.v, false, fb0.v,
                                                  (short)0, acc0, false, false);
    acc1 = __builtin_amdgcn_wmma_f32_16x16x32_f16(false, fa.v, false, fb1.v,
                                                  (short)0, acc1, false, false);
    __syncthreads();
  }

  // ---- epilogue: C/D layout -> lane holds col N=lane&15, rows i + 8*(lane>>4)
  v8f accs[2] = {acc0, acc1};
#pragma unroll
  for (int t = 0; t < 2; t++) {
    const int colTile = colTile0 + t * 16;
#pragma unroll
    for (int i = 0; i < 8; i++) {
      int r = rowBase + rowTile + i + 8 * (lane >> 4);
      int c = colBase + colTile + (lane & 15);
      float val = accs[t][i];
      if constexpr (EPI == 0) {
        _Float16* Co = (_Float16*)p.C + (long long)bat * p.sC;
        if (p.bias) val += p.bias[c];
        Co[(long long)r * p.ldc + c] = (_Float16)val;
      } else if constexpr (EPI == 1) {
        _Float16* Co = (_Float16*)p.C + (long long)bat * p.sC;
        Co[(long long)c * p.ldc + r] = (_Float16)val;
      } else if constexpr (EPI == 2) {
        val += p.bias[c];
        int b = r / p.Np, n = r % p.Np;
        int h = c >> 6, d = c & 63;
        _Float16* Co = (_Float16*)p.C;
        Co[(((long long)b * p.Hdim + h) * p.Np + n) * 64 + d] = (_Float16)val;
      } else if constexpr (EPI == 3) {
        float* Co = (float*)p.C;
        Co[(long long)r * p.ldc + c] = val + p.bias[c];
      } else if constexpr (EPI == 4) {
        float* Co = (float*)p.C + (long long)bat * p.sC;
        atomicAdd(&Co[(long long)r * p.ldc + c], val);
      } else { // 5: combine
        int b = bat >> 3, h = bat & 7;
        _Float16* Co = (_Float16*)p.C;
        Co[((long long)b * p.Np + r) * (p.Hdim * 64) + h * 64 + c] = (_Float16)val;
      }
    }
  }
}

// ---------------------------------------------------------------------------
// Fused routing: temperature MLP + logits + gumbel softmax + norm reduction.
// One wave per token (8 tokens / block). grid.x = B*H*(N/8).
// ---------------------------------------------------------------------------
__global__ __launch_bounds__(256) void route_kernel(
    const _Float16* __restrict__ qmid, const float* __restrict__ u,
    const float* __restrict__ W1, const float* __restrict__ b1,
    const float* __restrict__ W2, const float* __restrict__ b2,
    const float* __restrict__ biasT, const float* __restrict__ Ws,
    const float* __restrict__ bs, _Float16* __restrict__ slice_w,
    float* __restrict__ normOut) {
  __shared__ _Float16 qrow[8][64];
  __shared__ float lnorm[64];
  const int tid = threadIdx.x, wid = tid >> 5, lane = tid & 31;
  if (tid < 64) lnorm[tid] = 0.f;

  const int bh  = blockIdx.x >> 10;  // / (N/8)
  const int nch = blockIdx.x & 1023;
  const int h   = bh & 7;
  const long long tokBase = ((long long)bh * N_TOK + nch * 8 + wid) * 64;

  const _Float16* qr = qmid + tokBase;
  if (lane < 16) {  // 16 lanes x 4 halves = 64 halves, b64 traffic
    *(H4*)&qrow[wid][lane * 4] = *(const H4*)(qr + lane * 4);
  }
  __syncthreads();

  // hidden = gelu(q @ W1 + b1); temp = gelu(hidden @ W2 + b2) + bias[h]
  float h1a = b1[lane], h1b = b1[lane + 32];
  float la = bs[lane], lb = bs[lane + 32];
#pragma unroll 4
  for (int d = 0; d < 64; d++) {
    float qv = (float)qrow[wid][d];
    h1a = fmaf(qv, W1[d * 64 + lane], h1a);
    h1b = fmaf(qv, W1[d * 64 + lane + 32], h1b);
    la = fmaf(qv, Ws[d * 64 + lane], la);
    lb = fmaf(qv, Ws[d * 64 + lane + 32], lb);
  }
  h1a = gelu_f(h1a);
  h1b = gelu_f(h1b);
  float tacc = h1a * W2[lane] + h1b * W2[lane + 32];
#pragma unroll
  for (int o = 16; o; o >>= 1) tacc += __shfl_xor(tacc, o, 32);
  float temp = gelu_f(tacc + b2[0]) + biasT[h];
  temp = fmaxf(temp, 0.01f);

  const float* ur = u + tokBase;  // u is [B,H,N,G], same layout as qmid
  float ga = -logf(-logf(ur[lane] + 1e-8f) + 1e-8f);
  float gb = -logf(-logf(ur[lane + 32] + 1e-8f) + 1e-8f);
  float va = (la + ga) / temp, vb = (lb + gb) / temp;

  float mx = fmaxf(va, vb);
#pragma unroll
  for (int o = 16; o; o >>= 1) mx = fmaxf(mx, __shfl_xor(mx, o, 32));
  float ea = __expf(va - mx), eb = __expf(vb - mx);
  float s = ea + eb;
#pragma unroll
  for (int o = 16; o; o >>= 1) s += __shfl_xor(s, o, 32);
  float inv = 1.f / s;
  float wa = ea * inv, wb = eb * inv;

  _Float16* wr = slice_w + tokBase;
  wr[lane] = (_Float16)wa;
  wr[lane + 32] = (_Float16)wb;
  atomicAdd(&lnorm[lane], wa);
  atomicAdd(&lnorm[lane + 32], wb);
  __syncthreads();
  if (tid < 64) atomicAdd(&normOut[(long long)bh * 64 + tid], lnorm[tid]);
}

// ---------------------------------------------------------------------------
// Row softmax (in-place, f16): one wave per row of length rowLen.
// ---------------------------------------------------------------------------
__global__ __launch_bounds__(256) void softmax_rows(_Float16* s, int rowLen,
                                                    float scale) {
  const int row = blockIdx.x * 8 + (threadIdx.x >> 5);
  const int lane = threadIdx.x & 31;
  _Float16* p = s + (long long)row * rowLen;
  float mx = -1e30f;
  for (int i = lane; i < rowLen; i += 32)
    mx = fmaxf(mx, (float)p[i] * scale);
#pragma unroll
  for (int o = 16; o; o >>= 1) mx = fmaxf(mx, __shfl_xor(mx, o, 32));
  float sum = 0.f;
  for (int i = lane; i < rowLen; i += 32)
    sum += __expf((float)p[i] * scale - mx);
#pragma unroll
  for (int o = 16; o; o >>= 1) sum += __shfl_xor(sum, o, 32);
  float inv = 1.f / sum;
  for (int i = lane; i < rowLen; i += 32)
    p[i] = (_Float16)(__expf((float)p[i] * scale - mx) * inv);
}

__global__ void f32_to_f16(const float* __restrict__ src,
                           _Float16* __restrict__ dst, int n) {
  int i = blockIdx.x * 256 + threadIdx.x;
  if (i < n) dst[i] = (_Float16)src[i];
}

// transpose-convert: src f32 [K][N] row-major -> dst f16 [N][K]
__global__ void f32_to_f16_T(const float* __restrict__ src,
                             _Float16* __restrict__ dst, int K, int N) {
  int i = blockIdx.x * 256 + threadIdx.x;
  if (i < K * N) {
    int k = i / N, n = i % N;
    dst[(long long)n * K + k] = (_Float16)src[i];
  }
}

// ---------------------------------------------------------------------------
extern "C" void kernel_launch(void* const* d_in, const int* in_sizes, int n_in,
                              void* d_out, int out_size, void* d_ws,
                              size_t ws_size, hipStream_t stream) {
  (void)in_sizes; (void)n_in; (void)out_size; (void)ws_size;
  const float* q    = (const float*)d_in[0];
  const float* kv   = (const float*)d_in[1];
  const float* u    = (const float*)d_in[2];
  const float* Wq   = (const float*)d_in[3];
  const float* bq   = (const float*)d_in[4];
  const float* Wkv  = (const float*)d_in[5];
  const float* bkv  = (const float*)d_in[6];
  const float* W1   = (const float*)d_in[7];
  const float* b1   = (const float*)d_in[8];
  const float* W2   = (const float*)d_in[9];
  const float* b2   = (const float*)d_in[10];
  const float* biasT= (const float*)d_in[11];
  const float* Ws   = (const float*)d_in[12];
  const float* bs   = (const float*)d_in[13];
  const float* Wtq  = (const float*)d_in[14];
  const float* Wtk  = (const float*)d_in[15];
  const float* Wtv  = (const float*)d_in[16];
  const float* Wo   = (const float*)d_in[17];
  const float* bo   = (const float*)d_in[18];
  float* out = (float*)d_out;

  char* ws = (char*)d_ws;
  size_t off = 0;
  auto alloc = [&](size_t b) {
    size_t o = off;
    off += (b + 255) & ~(size_t)255;
    return o;
  };
  _Float16* Wq_t   = (_Float16*)(ws + alloc(512 * 512 * 2));   // [N][K]
  _Float16* Wkv_t  = (_Float16*)(ws + alloc(512 * 512 * 2));
  _Float16* Wo_t   = (_Float16*)(ws + alloc(512 * 512 * 2));
  _Float16* Wtq_t  = (_Float16*)(ws + alloc(64 * 64 * 2));
  _Float16* Wtk_t  = (_Float16*)(ws + alloc(64 * 64 * 2));
  _Float16* Wtv_t  = (_Float16*)(ws + alloc(64 * 64 * 2));
  _Float16* qmid_h = (_Float16*)(ws + alloc((size_t)B_ * H_ * N_TOK * D_ * 2));
  _Float16* kvmid_h= (_Float16*)(ws + alloc((size_t)B_ * H_ * M_TOK * D_ * 2));
  _Float16* slice_w= (_Float16*)(ws + alloc((size_t)B_ * H_ * N_TOK * G_ * 2));
  float*    normb  = (float*)   (ws + alloc((size_t)B_ * H_ * G_ * 4));
  float*    q_slice= (float*)   (ws + alloc((size_t)B_ * H_ * G_ * D_ * 4));
  _Float16* qs_h   = (_Float16*)(ws + alloc((size_t)B_ * H_ * G_ * D_ * 2));
  _Float16* ks_h   = (_Float16*)(ws + alloc((size_t)B_ * H_ * M_TOK * D_ * 2));
  _Float16* vsT_h  = (_Float16*)(ws + alloc((size_t)B_ * H_ * D_ * M_TOK * 2));
  _Float16* sc_h   = (_Float16*)(ws + alloc((size_t)B_ * H_ * G_ * M_TOK * 2));
  _Float16* osT_h  = (_Float16*)(ws + alloc((size_t)B_ * H_ * D_ * G_ * 2));
  _Float16* oflat_h= (_Float16*)(ws + alloc((size_t)B_ * N_TOK * DIM_ * 2));

  // weight conversions (all transposed to [N][K] for TDM-staged B operand)
  f32_to_f16_T<<<(512 * 512 + 255) / 256, 256, 0, stream>>>(Wq, Wq_t, 512, 512);
  f32_to_f16_T<<<(512 * 512 + 255) / 256, 256, 0, stream>>>(Wkv, Wkv_t, 512, 512);
  f32_to_f16_T<<<(512 * 512 + 255) / 256, 256, 0, stream>>>(Wo, Wo_t, 512, 512);
  f32_to_f16_T<<<(64 * 64 + 255) / 256, 256, 0, stream>>>(Wtq, Wtq_t, 64, 64);
  f32_to_f16_T<<<(64 * 64 + 255) / 256, 256, 0, stream>>>(Wtk, Wtk_t, 64, 64);
  f32_to_f16_T<<<(64 * 64 + 255) / 256, 256, 0, stream>>>(Wtv, Wtv_t, 64, 64);
  (void)hipMemsetAsync(normb, 0, (size_t)B_ * H_ * G_ * 4, stream);
  (void)hipMemsetAsync(q_slice, 0, (size_t)B_ * H_ * G_ * D_ * 4, stream);

  GP p{};
  // 1) q in-projection -> qmid [b,h,n,d] f16
  p = GP{}; p.A = q; p.B = Wq_t; p.bias = bq; p.C = qmid_h;
  p.M = B_ * N_TOK; p.N = DIM_; p.K = DIM_; p.lda = DIM_; p.ldb = DIM_;
  p.ksplit = 1; p.Np = N_TOK; p.Hdim = H_;
  gemm_wmma<0, 1, 2><<<dim3(8, 512, 1), 256, 0, stream>>>(p);

  // 2) kv in-projection -> kvmid [b,h,m,d] f16
  p = GP{}; p.A = kv; p.B = Wkv_t; p.bias = bkv; p.C = kvmid_h;
  p.M = B_ * M_TOK; p.N = DIM_; p.K = DIM_; p.lda = DIM_; p.ldb = DIM_;
  p.ksplit = 1; p.Np = M_TOK; p.Hdim = H_;
  gemm_wmma<0, 1, 2><<<dim3(8, 64, 1), 256, 0, stream>>>(p);

  // 3) fused routing: slice_w f16 + norm f32
  route_kernel<<<B_ * H_ * (N_TOK / 8), 256, 0, stream>>>(
      qmid_h, u, W1, b1, W2, b2, biasT, Ws, bs, slice_w, normb);

  // 4) dispatch: q_slice[g,d] += sum_n w[n,g]*qmid[n,d], K split 8-way
  p = GP{}; p.A = slice_w; p.B = qmid_h; p.C = q_slice;
  p.M = G_; p.N = D_; p.K = N_TOK; p.lda = G_; p.ldb = D_; p.ldc = D_;
  p.sA = (long long)N_TOK * G_; p.sB = (long long)N_TOK * D_; p.sC = G_ * D_;
  p.ksplit = 8; p.Np = N_TOK; p.Hdim = H_;
  gemm_wmma<3, 0, 4><<<dim3(1, 1, B_ * H_ * 8), 256, 0, stream>>>(p);

  // 5) qs = (q_slice / (norm+1e-5)) @ Wtq -> f16
  p = GP{}; p.A = q_slice; p.B = Wtq_t; p.rowScale = normb; p.C = qs_h;
  p.M = G_; p.N = D_; p.K = D_; p.lda = D_; p.ldb = D_; p.ldc = D_;
  p.sA = G_ * D_; p.sB = 0; p.sC = G_ * D_;
  p.ksplit = 1; p.Np = N_TOK; p.Hdim = H_;
  gemm_wmma<2, 1, 0><<<dim3(1, 1, B_ * H_), 256, 0, stream>>>(p);

  // 6) ks = kvmid @ Wtk (flat 32768x64x64) -> f16
  p = GP{}; p.A = kvmid_h; p.B = Wtk_t; p.C = ks_h;
  p.M = B_ * H_ * M_TOK; p.N = D_; p.K = D_; p.lda = D_; p.ldb = D_; p.ldc = D_;
  p.ksplit = 1; p.Np = M_TOK; p.Hdim = H_;
  gemm_wmma<1, 1, 0><<<dim3(1, 512, 1), 256, 0, stream>>>(p);

  // 7) vsT = (kvmid @ Wtv)^T per (b,h) -> [b,h,d,m] f16
  p = GP{}; p.A = kvmid_h; p.B = Wtv_t; p.C = vsT_h;
  p.M = M_TOK; p.N = D_; p.K = D_; p.lda = D_; p.ldb = D_; p.ldc = M_TOK;
  p.sA = (long long)M_TOK * D_; p.sB = 0; p.sC = (long long)D_ * M_TOK;
  p.ksplit = 1; p.Np = M_TOK; p.Hdim = H_;
  gemm_wmma<1, 1, 1><<<dim3(1, 16, B_ * H_), 256, 0, stream>>>(p);

  // 8) scores = qs @ ks^T per (b,h): [g,m] f16 (ks is [m][d] = B col-major)
  p = GP{}; p.A = qs_h; p.B = ks_h; p.C = sc_h;
  p.M = G_; p.N = M_TOK; p.K = D_; p.lda = D_; p.ldb = D_; p.ldc = M_TOK;
  p.sA = G_ * D_; p.sB = (long long)M_TOK * D_; p.sC = (long long)G_ * M_TOK;
  p.ksplit = 1; p.Np = M_TOK; p.Hdim = H_;
  gemm_wmma<1, 1, 0><<<dim3(16, 1, B_ * H_), 256, 0, stream>>>(p);

  // 9) softmax over m with scale D^-0.5
  softmax_rows<<<(B_ * H_ * G_) / 8, 256, 0, stream>>>(sc_h, M_TOK, 0.125f);

  // 10) out_sliceT = (P @ vs)^T per (b,h) -> [b,h,d,g] f16
  p = GP{}; p.A = sc_h; p.B = vsT_h; p.C = osT_h;
  p.M = G_; p.N = D_; p.K = M_TOK; p.lda = M_TOK; p.ldb = M_TOK; p.ldc = G_;
  p.sA = (long long)G_ * M_TOK; p.sB = (long long)D_ * M_TOK; p.sC = D_ * G_;
  p.ksplit = 1; p.Np = M_TOK; p.Hdim = H_;
  gemm_wmma<1, 1, 1><<<dim3(1, 1, B_ * H_), 256, 0, stream>>>(p);

  // 11) combine: out_flat[b,n,h*64+d] = sum_g w[n,g]*out_slice[g,d]
  p = GP{}; p.A = slice_w; p.B = osT_h; p.C = oflat_h;
  p.M = N_TOK; p.N = D_; p.K = G_; p.lda = G_; p.ldb = G_;
  p.sA = (long long)N_TOK * G_; p.sB = D_ * G_;
  p.ksplit = 1; p.Np = N_TOK; p.Hdim = H_;
  gemm_wmma<1, 1, 5><<<dim3(1, 128, B_ * H_), 256, 0, stream>>>(p);

  // 12) final: out = out_flat @ Wo + bo (f32 output)
  p = GP{}; p.A = oflat_h; p.B = Wo_t; p.bias = bo; p.C = out;
  p.M = B_ * N_TOK; p.N = DIM_; p.K = DIM_; p.lda = DIM_; p.ldb = DIM_;
  p.ldc = DIM_; p.ksplit = 1; p.Np = N_TOK; p.Hdim = H_;
  gemm_wmma<1, 1, 3><<<dim3(8, 512, 1), 256, 0, stream>>>(p);
}
